// HEALPixAttention_609885356096
// MI455X (gfx1250) — matrix operand order repaired and checked by reading, not compiled, not run on hardware
//
#include <hip/hip_runtime.h>
#include <hip/hip_bf16.h>
#include <math.h>

typedef __attribute__((ext_vector_type(16))) __bf16 v16bf;
typedef __attribute__((ext_vector_type(2)))  __bf16 v2bf;
typedef __attribute__((ext_vector_type(8)))  float  v8f;

#define B_  8
#define N_  1024
#define D_  512
#define H_  8
#define HD_ 64

static __device__ __forceinline__ v8f wmma_bf16f32(v16bf a, v16bf b, v8f c) {
  // D = A(16x32) * B(32x16) + C, f32 accumulate
  return __builtin_amdgcn_wmma_f32_16x16x32_bf16(false, a, false, b, (short)0, c, false, false);
}

static __device__ __forceinline__ v8f vzero8() {
  v8f z = {0.f, 0.f, 0.f, 0.f, 0.f, 0.f, 0.f, 0.f};
  return z;
}

// Pack two f32 into one dword of bf16x2 (lowers to v_cvt_pk_bf16_f32).
static __device__ __forceinline__ unsigned pk_bf16(float a, float b) {
  v2bf t; t[0] = (__bf16)a; t[1] = (__bf16)b;
  return __builtin_bit_cast(unsigned, t);
}

// ---------------------------------------------------------------------------
// Kernel 1: qkv = x @ W_qkv + b_qkv ; split into Q (pre-scaled 1/8), K, V^T
// Block tile 128(M) x 64(N), K-step 32, 8 waves each doing a 32x32 subtile.
// ---------------------------------------------------------------------------
__global__ __launch_bounds__(256, 2)
void qkv_proj_kernel(const float* __restrict__ x, const float* __restrict__ Wqkv,
                     const float* __restrict__ bqkv,
                     __bf16* __restrict__ Qw, __bf16* __restrict__ Kw,
                     __bf16* __restrict__ Vtw)
{
  __shared__ __bf16 ldsA[128][40];   // row-major A tile (pad 40 -> conflict free)
  __shared__ __bf16 ldsBT[64][40];   // B tile stored transposed: [n][k]

  const int tid  = threadIdx.x;
  const int lane = tid & 31;
  const int w    = tid >> 5;
  const int wrow = w >> 1;           // 0..3
  const int wcol = w & 1;            // 0..1
  const int m0   = blockIdx.x * 128;
  const int n0   = blockIdx.y * 64;
  const int nl   = lane & 15;
  const int mb8  = (lane >> 4) << 3; // 0 or 8  (C row base)
  const int kb16 = (lane >> 4) << 4; // 0 or 16 (B k base)

  v8f acc[2][2];
  acc[0][0] = vzero8(); acc[0][1] = vzero8();
  acc[1][0] = vzero8(); acc[1][1] = vzero8();

  for (int k0 = 0; k0 < D_; k0 += 32) {
    __syncthreads();
    // A tile: x[m0..m0+127][k0..k0+31], f32 -> packed bf16x2 dword stores
#pragma unroll
    for (int j = 0; j < 4; ++j) {
      int i = tid + j * 256;
      int r = i >> 3;
      int c = (i & 7) << 2;
      float4 v = *(const float4*)(x + (size_t)(m0 + r) * D_ + k0 + c);
      *(unsigned*)&ldsA[r][c]     = pk_bf16(v.x, v.y);
      *(unsigned*)&ldsA[r][c + 2] = pk_bf16(v.z, v.w);
    }
    // B tile: two adjacent k rows per thread so the transposed store packs
    // contiguously along k: ldsBT[n][k..k+1] = one dword.
    {
      int kp = tid >> 4;            // 0..15 -> k pair
      int c  = (tid & 15) << 2;     // 0..60
      const float* w0 = Wqkv + (size_t)(k0 + 2 * kp) * (3 * D_) + n0 + c;
      float4 va = *(const float4*)w0;
      float4 vb = *(const float4*)(w0 + 3 * D_);
      *(unsigned*)&ldsBT[c + 0][2 * kp] = pk_bf16(va.x, vb.x);
      *(unsigned*)&ldsBT[c + 1][2 * kp] = pk_bf16(va.y, vb.y);
      *(unsigned*)&ldsBT[c + 2][2 * kp] = pk_bf16(va.z, vb.z);
      *(unsigned*)&ldsBT[c + 3][2 * kp] = pk_bf16(va.w, vb.w);
    }
    __syncthreads();

    v16bf af[2], bfr[2];
#pragma unroll
    for (int ma = 0; ma < 2; ++ma) {
      const __bf16* p = &ldsA[wrow * 32 + ma * 16 + nl][0];
#pragma unroll
      for (int e = 0; e < 8; ++e) { af[ma][e] = p[kb16 + e]; af[ma][8 + e] = p[kb16 + 8 + e]; }
    }
#pragma unroll
    for (int nb = 0; nb < 2; ++nb) {
      const __bf16* p = &ldsBT[wcol * 32 + nb * 16 + nl][kb16];
#pragma unroll
      for (int e = 0; e < 16; ++e) bfr[nb][e] = p[e];
    }
#pragma unroll
    for (int ma = 0; ma < 2; ++ma)
#pragma unroll
      for (int nb = 0; nb < 2; ++nb)
        acc[ma][nb] = wmma_bf16f32(af[ma], bfr[nb], acc[ma][nb]);
  }

  // Branchless epilogue: per-lane select of target buffer / index / scale.
#pragma unroll
  for (int ma = 0; ma < 2; ++ma) {
#pragma unroll
    for (int nb = 0; nb < 2; ++nb) {
      const int col = n0 + wcol * 32 + nb * 16 + nl;
      const int h  = col / 192;
      const int rm = col - h * 192;
      const int s  = rm >> 6;                 // 0=q 1=k 2=v
      const int t  = rm & 63;
      const float bb    = bqkv[col];
      const float scale = (s == 0) ? 0.125f : 1.0f;   // fold 1/sqrt(64) into Q
      __bf16* base = (s == 0) ? Qw : ((s == 1) ? Kw : Vtw);
#pragma unroll
      for (int r = 0; r < 8; ++r) {
        const int row = m0 + wrow * 32 + ma * 16 + mb8 + r;
        const int bi  = row >> 10;            // batch
        const int q   = row & (N_ - 1);       // token
        const size_t head   = (size_t)(bi * H_ + h);
        const size_t idx_qk = (head * N_ + q) * HD_ + t;   // [b,h,n,t]
        const size_t idx_v  = (head * HD_ + t) * N_ + q;   // [b,h,t,n] (V^T)
        const size_t idx = (s == 2) ? idx_v : idx_qk;
        base[idx] = (__bf16)((acc[ma][nb][r] + bb) * scale);
      }
    }
  }
}

// ---------------------------------------------------------------------------
// Kernel 2: flash attention. Block = (b, 16-query tile); 8 waves = 8 heads.
// Bias tile DMA'd to LDS with GLOBAL_LOAD_ASYNC_TO_LDS_B128 (ASYNCcnt),
// online softmax, P re-layout through wave-private LDS, O += P*V via WMMA.
// ---------------------------------------------------------------------------
__global__ __launch_bounds__(256, 2)
void attn_kernel(const __bf16* __restrict__ Qw, const __bf16* __restrict__ Kw,
                 const __bf16* __restrict__ Vtw, const float* __restrict__ rbias,
                 __bf16* __restrict__ yw)
{
  __shared__ float  ldsBias[16 * 32 * 8];   // [q][k][h], 16KB
  __shared__ __bf16 ldsP[8][16][32];        // per-wave P staging, 8KB

  const int tid  = threadIdx.x;
  const int lane = tid & 31;
  const int h    = tid >> 5;                // wave id == head
  const int b    = blockIdx.x >> 6;
  const int q0   = (blockIdx.x & 63) << 4;
  const int nl   = lane & 15;
  const int mb8  = (lane >> 4) << 3;        // C-layout row base
  const int kb16 = (lane >> 4) << 4;        // B-frag k base

  const size_t headOff = ((size_t)(b * H_ + h)) * N_ * HD_;

  // Q fragments (16 queries x 64 dims = two 16x32 A-frags), already scaled.
  v16bf qa[2];
  {
    const __bf16* pq = Qw + headOff + (size_t)(q0 + nl) * HD_;
#pragma unroll
    for (int f = 0; f < 2; ++f)
#pragma unroll
      for (int e = 0; e < 8; ++e) {
        qa[f][e]     = pq[f * 32 + kb16 + e];
        qa[f][8 + e] = pq[f * 32 + kb16 + 8 + e];
      }
  }

  float mst[8], lst[8];
#pragma unroll
  for (int r = 0; r < 8; ++r) { mst[r] = -INFINITY; lst[r] = 0.f; }
  v8f o[4];
#pragma unroll
  for (int c = 0; c < 4; ++c) o[c] = vzero8();

  for (int kb = 0; kb < N_; kb += 32) {
    __syncthreads();   // previous iteration's ldsBias reads complete
    // Async DMA bias[b, q0:q0+16, kb:kb+32, 0:8] -> LDS, no VGPR round-trip.
    // 1024 x b128 transfers, 4 per thread, fully coalesced.
#pragma unroll
    for (int j = 0; j < 4; ++j) {
      int i = tid + j * 256;
      int q = i >> 6;
      int rem = (i & 63) << 2;
      const float* gp = rbias + (((size_t)b * N_ + q0 + q) * N_ + kb) * 8 + rem;
      unsigned ldsaddr = (unsigned)(size_t)&ldsBias[q * 256 + rem];
      asm volatile("global_load_async_to_lds_b128 %0, %1, off"
                   :: "v"(ldsaddr), "v"((unsigned long long)(size_t)gp)
                   : "memory");
    }
    // Prefetch next K / V^T tiles while the DMA runs.
    if (kb + 32 < N_) {
      __builtin_prefetch(Kw + headOff + (size_t)(kb + 32 + nl) * HD_, 0, 0);
      __builtin_prefetch(Vtw + headOff + (size_t)nl * N_ + kb + 32, 0, 0);
    }
    asm volatile("s_wait_asynccnt 0x0" ::: "memory");
    __syncthreads();

    // S = Q * K^T + bias : two 16x16 key subtiles
    v8f s[2];
#pragma unroll
    for (int t = 0; t < 2; ++t) {
      const __bf16* pk = Kw + headOff + (size_t)(kb + t * 16 + nl) * HD_;
      v16bf k0f, k1f;
#pragma unroll
      for (int e = 0; e < 16; ++e) { k0f[e] = pk[kb16 + e]; k1f[e] = pk[32 + kb16 + e]; }
      v8f sv = wmma_bf16f32(qa[0], k0f, vzero8());
      sv = wmma_bf16f32(qa[1], k1f, sv);
#pragma unroll
      for (int r = 0; r < 8; ++r)
        sv[r] += ldsBias[((mb8 + r) * 32 + t * 16 + nl) * 8 + h];
      s[t] = sv;
    }

    // Online softmax (row stats across 16 lanes of each half-wave)
    float al[8];
#pragma unroll
    for (int r = 0; r < 8; ++r) {
      float mx = fmaxf(s[0][r], s[1][r]);
#pragma unroll
      for (int msk = 1; msk < 16; msk <<= 1) mx = fmaxf(mx, __shfl_xor(mx, msk, 16));
      float nm = fmaxf(mst[r], mx);
      al[r] = __expf(mst[r] - nm);
      float p0 = __expf(s[0][r] - nm);
      float p1 = __expf(s[1][r] - nm);
      s[0][r] = p0; s[1][r] = p1;
      float sum = p0 + p1;
#pragma unroll
      for (int msk = 1; msk < 16; msk <<= 1) sum += __shfl_xor(sum, msk, 16);
      lst[r] = lst[r] * al[r] + sum;
      mst[r] = nm;
    }
#pragma unroll
    for (int c = 0; c < 4; ++c)
#pragma unroll
      for (int r = 0; r < 8; ++r) o[c][r] *= al[r];

    // P: C-layout -> A-layout through wave-private LDS
#pragma unroll
    for (int r = 0; r < 8; ++r) {
      int m = mb8 + r;
      ldsP[h][m][nl]      = (__bf16)s[0][r];
      ldsP[h][m][16 + nl] = (__bf16)s[1][r];
    }
    asm volatile("s_wait_dscnt 0" ::: "memory");
    v16bf pf;
    {
      const __bf16* pp = &ldsP[h][nl][0];
#pragma unroll
      for (int e = 0; e < 8; ++e) { pf[e] = pp[kb16 + e]; pf[8 + e] = pp[kb16 + 8 + e]; }
    }

    // O += P(16x32) * V(32x64) : V^T gives contiguous B-frag loads
#pragma unroll
    for (int c = 0; c < 4; ++c) {
      const __bf16* pv = Vtw + headOff + (size_t)(c * 16 + nl) * N_ + kb + kb16;
      v16bf vf;
#pragma unroll
      for (int e = 0; e < 16; ++e) vf[e] = pv[e];
      o[c] = wmma_bf16f32(pf, vf, o[c]);
    }
  }

  // Normalize + write y[b, q, h*64 + t] as bf16 for the output GEMM
#pragma unroll
  for (int c = 0; c < 4; ++c)
#pragma unroll
    for (int r = 0; r < 8; ++r) {
      int m = mb8 + r;
      float v = o[c][r] / lst[r];
      yw[((size_t)(b * N_) + q0 + m) * D_ + h * HD_ + c * 16 + nl] = (__bf16)v;
    }
}

// ---------------------------------------------------------------------------
// Kernel 3: out = y @ W_o + b_o  (f32 output)
// ---------------------------------------------------------------------------
__global__ __launch_bounds__(256, 2)
void out_proj_kernel(const __bf16* __restrict__ yw, const float* __restrict__ Wo,
                     const float* __restrict__ bo, float* __restrict__ out)
{
  __shared__ __bf16 ldsA[128][40];
  __shared__ __bf16 ldsBT[64][40];

  const int tid  = threadIdx.x;
  const int lane = tid & 31;
  const int w    = tid >> 5;
  const int wrow = w >> 1;
  const int wcol = w & 1;
  const int m0   = blockIdx.x * 128;
  const int n0   = blockIdx.y * 64;
  const int nl   = lane & 15;
  const int mb8  = (lane >> 4) << 3;
  const int kb16 = (lane >> 4) << 4;

  v8f acc[2][2];
  acc[0][0] = vzero8(); acc[0][1] = vzero8();
  acc[1][0] = vzero8(); acc[1][1] = vzero8();

  for (int k0 = 0; k0 < D_; k0 += 32) {
    __syncthreads();
    // A tile: already bf16, 16B vector copies (pad 40 keeps 16B alignment)
#pragma unroll
    for (int j = 0; j < 2; ++j) {
      int i = tid + j * 256;
      int r = i >> 2;
      int c = (i & 3) << 3;
      *(uint4*)&ldsA[r][c] = *(const uint4*)(yw + (size_t)(m0 + r) * D_ + k0 + c);
    }
    // B tile: two adjacent k rows -> packed bf16x2 stores along k
    {
      int kp = tid >> 4;
      int c  = (tid & 15) << 2;
      const float* w0 = Wo + (size_t)(k0 + 2 * kp) * D_ + n0 + c;
      float4 va = *(const float4*)w0;
      float4 vb = *(const float4*)(w0 + D_);
      *(unsigned*)&ldsBT[c + 0][2 * kp] = pk_bf16(va.x, vb.x);
      *(unsigned*)&ldsBT[c + 1][2 * kp] = pk_bf16(va.y, vb.y);
      *(unsigned*)&ldsBT[c + 2][2 * kp] = pk_bf16(va.z, vb.z);
      *(unsigned*)&ldsBT[c + 3][2 * kp] = pk_bf16(va.w, vb.w);
    }
    __syncthreads();

    v16bf af[2], bfr[2];
#pragma unroll
    for (int ma = 0; ma < 2; ++ma) {
      const __bf16* p = &ldsA[wrow * 32 + ma * 16 + nl][0];
#pragma unroll
      for (int e = 0; e < 8; ++e) { af[ma][e] = p[kb16 + e]; af[ma][8 + e] = p[kb16 + 8 + e]; }
    }
#pragma unroll
    for (int nb = 0; nb < 2; ++nb) {
      const __bf16* p = &ldsBT[wcol * 32 + nb * 16 + nl][kb16];
#pragma unroll
      for (int e = 0; e < 16; ++e) bfr[nb][e] = p[e];
    }
#pragma unroll
    for (int ma = 0; ma < 2; ++ma)
#pragma unroll
      for (int nb = 0; nb < 2; ++nb)
        acc[ma][nb] = wmma_bf16f32(af[ma], bfr[nb], acc[ma][nb]);
  }

#pragma unroll
  for (int ma = 0; ma < 2; ++ma)
#pragma unroll
    for (int nb = 0; nb < 2; ++nb) {
      const int col = n0 + wcol * 32 + nb * 16 + nl;
      const float bb = bo[col];
#pragma unroll
      for (int r = 0; r < 8; ++r) {
        const int row = m0 + wrow * 32 + ma * 16 + mb8 + r;
        out[(size_t)row * D_ + col] = acc[ma][nb][r] + bb;
      }
    }
}

// ---------------------------------------------------------------------------
extern "C" void kernel_launch(void* const* d_in, const int* in_sizes, int n_in,
                              void* d_out, int out_size, void* d_ws, size_t ws_size,
                              hipStream_t stream)
{
  (void)in_sizes; (void)n_in; (void)out_size; (void)ws_size;
  const float* x     = (const float*)d_in[0];
  const float* rbias = (const float*)d_in[1];
  const float* Wqkv  = (const float*)d_in[2];
  const float* bqkv  = (const float*)d_in[3];
  const float* Wo    = (const float*)d_in[4];
  const float* bo    = (const float*)d_in[5];
  float* out = (float*)d_out;

  char* ws = (char*)d_ws;
  const size_t SZ = (size_t)B_ * H_ * N_ * HD_ * sizeof(__bf16);  // 8 MB
  __bf16* Qw  = (__bf16*)(ws);
  __bf16* Kw  = (__bf16*)(ws + SZ);
  __bf16* Vtw = (__bf16*)(ws + 2 * SZ);
  __bf16* yw  = (__bf16*)(ws + 3 * SZ);

  qkv_proj_kernel<<<dim3(64, 24), 256, 0, stream>>>(x, Wqkv, bqkv, Qw, Kw, Vtw);
  attn_kernel<<<dim3(512), 256, 0, stream>>>(Qw, Kw, Vtw, rbias, yw);
  out_proj_kernel<<<dim3(64, 8), 256, 0, stream>>>(yw, Wo, bo, out);
}
